// Head_21440476741642
// MI455X (gfx1250) — matrix operand order repaired and checked by reading, not compiled, not run on hardware
//
#include <hip/hip_runtime.h>
#include <hip/hip_bf16.h>

// ---------------------------------------------------------------------------
// Fused causal single-head attention for MI455X (gfx1250, wave32, WMMA).
//   Phase 1: q/k/v projections (x @ W) in bf16 WMMA, f32 accumulate.
//            v stored transposed (h-major) so phase 2 gets contiguous B frags.
//   Phase 2: flash-attention over causal key blocks, bf16 WMMA + online
//            softmax (exp2-folded), f32 output.
// ---------------------------------------------------------------------------

typedef __attribute__((ext_vector_type(16))) __bf16 v16bf;
typedef __attribute__((ext_vector_type(8)))  __bf16 v8bf;
typedef __attribute__((ext_vector_type(2)))  __bf16 v2bf;
typedef __attribute__((ext_vector_type(16))) float  v16f;
typedef __attribute__((ext_vector_type(8)))  float  v8f;
typedef __attribute__((ext_vector_type(4)))  float  v4f;
typedef __attribute__((ext_vector_type(2)))  float  v2f;

#define N_EMBD 768
#define HEAD   128
#define TSEQ   1024
#define BATCH  16
#define NROWS  (BATCH * TSEQ)   // 16384

union Frag16 {           // register-aliasing views of one A/B fragment
  v16bf v;
  v8bf  h[2];
};
union F32x16 {           // 16 floats as four 16B loads
  v16f v;
  v4f  q[4];
};
union BF8 {              // packed v8bf viewed as dwords (for paired stores)
  v8bf     v;
  unsigned u[4];
};
union BF2 {              // packed bf16 pair as one dword
  v2bf     v;
  unsigned u;
};

__device__ __forceinline__ float fast_exp2(float x) {
#if __has_builtin(__builtin_amdgcn_exp2f)
  return __builtin_amdgcn_exp2f(x);
#else
  return __expf(x * 0.6931471805599453f);
#endif
}

// ---------------------------------------------------------------------------
// Kernel 1: out = x @ W for one of {Wq, Wk, Wv}, 128-row tile per block.
// A frag (16x32 bf16): lane l holds row (l&15); element i -> k =
//   (l>>4)*16 + i  within the 32-wide K chunk.
// B frag (32x16 bf16): lane l holds col (l&15); element i -> k =
//   (l>>4)*16 + i.
// ---------------------------------------------------------------------------
__global__ __launch_bounds__(256) void qkv_proj_kernel(
    const float* __restrict__ x,
    const float* __restrict__ Wq, const float* __restrict__ Wk,
    const float* __restrict__ Wv,
    __bf16* __restrict__ qws, __bf16* __restrict__ kws,
    __bf16* __restrict__ vTws)
{
  __shared__ __align__(32) __bf16 ldsB[8 * 32 * 16];  // 8 N-tiles of B frags

  const int t    = threadIdx.x;
  const int wv   = t >> 5;        // wave 0..7 -> 16-row strip
  const int l    = t & 31;        // lane
  const int l15  = l & 15;
  const int lh   = l >> 4;
  const int row0 = blockIdx.x * 128;
  const int mat  = blockIdx.y;    // 0=q, 1=k, 2=v(transposed)
  const float* W = (mat == 0) ? Wq : (mat == 1) ? Wk : Wv;

  v8f acc[8] = {};  // 16 rows x 128 cols per wave

  const int    arow = row0 + wv * 16 + l15;
  const float* xrow = x + (size_t)arow * N_EMBD;

  for (int kc = 0; kc < N_EMBD / 32; ++kc) {
    const int k0 = kc * 32;
    __syncthreads();  // previous iteration's LDS reads complete

    // --- stage W[k0:k0+32, 0:128] into LDS in B-fragment order ---
    // Consecutive k's of one column are adjacent in a fragment: convert
    // (W[k][c], W[k+1][c]) pairs with v_cvt_pk_bf16_f32 and store dwords.
#pragma unroll
    for (int it = 0; it < 2; ++it) {
      const int g   = it * 256 + t;        // 0..511 pair-tasks
      const int kl0 = (g >> 5) * 2;        // even k within 32-chunk
      const int col = (g & 31) * 4;        // column quad
      const v4f wa = *(const v4f*)&W[(size_t)(k0 + kl0) * HEAD + col];
      const v4f wb = *(const v4f*)&W[(size_t)(k0 + kl0 + 1) * HEAD + col];
      const int nt    = col >> 4;
      const int lbase = (kl0 >= 16) ? 16 : 0;
      const int ib    = kl0 & 15;          // even -> dword aligned
#pragma unroll
      for (int j = 0; j < 4; ++j) {
        v2f tf; tf[0] = wa[j]; tf[1] = wb[j];
        BF2 pr;
        pr.v = __builtin_convertvector(tf, v2bf);
        const int lb = ((col + j) & 15) + lbase;
        *(unsigned*)&ldsB[(nt * 32 + lb) * 16 + ib] = pr.u;
      }
    }
    __syncthreads();

    // --- A fragment straight from global (one contiguous 64B run per lane) ---
    const float* xp = xrow + k0 + lh * 16;
    __builtin_prefetch(xp + 32, 0, 1);     // next k-chunk of this lane
    F32x16 u;
    u.q[0] = *(const v4f*)(xp + 0);
    u.q[1] = *(const v4f*)(xp + 4);
    u.q[2] = *(const v4f*)(xp + 8);
    u.q[3] = *(const v4f*)(xp + 12);
    const v16bf a = __builtin_convertvector(u.v, v16bf);

    // --- 8 WMMAs with B fragments double-buffered from LDS so each WMMA
    //     waits only on the older DS load (dscnt<=1), not a full drain ---
    const __bf16* bbase = &ldsB[(size_t)l * 16];
    v16bf bcur = *(const v16bf*)(bbase);
#pragma unroll
    for (int n = 0; n < 8; ++n) {
      v16bf bnext;
      if (n < 7) bnext = *(const v16bf*)(bbase + (size_t)(n + 1) * 32 * 16);
      acc[n] = __builtin_amdgcn_wmma_f32_16x16x32_bf16(
          false, a, false, bcur, (short)0, acc[n], false, false);
      if (n < 7) bcur = bnext;
    }
  }

  // --- epilogue: C/D layout -> lane l, vgpr r holds (row r + 8*(l>>4), col l&15)
  if (mat < 2) {
    __bf16* dst = (mat == 0) ? qws : kws;
#pragma unroll
    for (int n = 0; n < 8; ++n) {
      const v8bf pb = __builtin_convertvector(acc[n], v8bf);
#pragma unroll
      for (int r = 0; r < 8; ++r) {
        const int row = row0 + wv * 16 + r + lh * 8;
        dst[(size_t)row * HEAD + n * 16 + l15] = pb[r];
      }
    }
  } else {
    // v transposed: vT[b][h][t]; adjacent r -> adjacent t, store as dwords
#pragma unroll
    for (int n = 0; n < 8; ++n) {
      BF8 pb;
      pb.v = __builtin_convertvector(acc[n], v8bf);
#pragma unroll
      for (int j = 0; j < 4; ++j) {
        const int row = row0 + wv * 16 + 2 * j + lh * 8;  // even token id
        const int bb  = row >> 10;
        const int tt  = row & 1023;
        const int h   = n * 16 + l15;
        *(unsigned*)&vTws[((size_t)bb * HEAD + h) * TSEQ + tt] = pb.u[j];
      }
    }
  }
}

// ---------------------------------------------------------------------------
// Kernel 2: flash attention. Block = (q-block of 128 rows, batch).
// Each wave owns 16 query rows. Causal => only key blocks jb <= qb visited.
// ---------------------------------------------------------------------------
__global__ __launch_bounds__(256) void attn_kernel(
    const __bf16* __restrict__ q, const __bf16* __restrict__ k,
    const __bf16* __restrict__ vT, float* __restrict__ out)
{
  __shared__ __align__(32) __bf16 plds[8][16][128];  // per-wave P staging (32KB)

  const int t   = threadIdx.x;
  const int wv  = t >> 5;
  const int l   = t & 31;
  const int l15 = l & 15;
  const int lh  = l >> 4;
  const int qb  = blockIdx.x;       // query block (128 rows)
  const int b   = blockIdx.y;       // batch
  const int t0  = qb * 128;
  // softmax in base-2: exp(s*scale - m*scale) == exp2((s - m)*cexp)
  const float cexp = 0.03608439182435161f * 1.4426950408889634f;

  // --- preload Q A-fragments (4 chunks over head dim 128) ---
  const int qrow = t0 + wv * 16 + l15;
  const __bf16* qp = q + ((size_t)b * TSEQ + qrow) * HEAD;
  v16bf qa[4];
#pragma unroll
  for (int c = 0; c < 4; ++c) {
    const int hb = c * 32 + lh * 16;
    Frag16 u;
    u.h[0] = *(const v8bf*)(qp + hb);
    u.h[1] = *(const v8bf*)(qp + hb + 8);
    qa[c] = u.v;
  }

  v8f   o[8] = {};
  float mrow[8], lrow[8];   // raw-score row max, exp2-domain row sum
#pragma unroll
  for (int r = 0; r < 8; ++r) { mrow[r] = -__builtin_inff(); lrow[r] = 0.0f; }

  const __bf16* kbat = k  + (size_t)b * TSEQ * HEAD;
  const __bf16* vbat = vT + (size_t)b * HEAD * TSEQ;

  for (int jb = 0; jb <= qb; ++jb) {
    // ---- prefetch next key block (each lane-half covers one 128B line) ----
    if (jb + 1 <= qb) {
#pragma unroll
      for (int n = 0; n < 8; ++n) {
        __builtin_prefetch(kbat + (size_t)((jb + 1) * 128 + n * 16 + l15) * HEAD
                                + lh * 64, 0, 3);
        __builtin_prefetch(vbat + (size_t)(n * 16 + l15) * TSEQ
                                + (jb + 1) * 128 + lh * 64, 0, 3);
      }
    }

    // ---- S = (Q K^T) for this 128-key block: 8 tiles of 16x16 ----
    v8f s[8];
#pragma unroll
    for (int n = 0; n < 8; ++n) {
      v8f sa = {};
      const int krow = jb * 128 + n * 16 + l15;       // B frag col = key
      const __bf16* kp = kbat + (size_t)krow * HEAD;  // contraction over h
#pragma unroll
      for (int c = 0; c < 4; ++c) {
        // B-fragment = 32 contiguous bytes of this key's h-vector
        const v16bf bf = *(const v16bf*)(kp + c * 32 + lh * 16);
        sa = __builtin_amdgcn_wmma_f32_16x16x32_bf16(
            false, qa[c], false, bf, (short)0, sa, false, false);
      }
      s[n] = sa;
    }

    // ---- causal mask in raw-score domain (diagonal block only) ----
    if (jb == qb) {
#pragma unroll
      for (int n = 0; n < 8; ++n) {
        const int col = jb * 128 + n * 16 + l15;
#pragma unroll
        for (int r = 0; r < 8; ++r) {
          const int row = t0 + wv * 16 + r + lh * 8;
          s[n][r] = (col <= row) ? s[n][r] : -__builtin_inff();
        }
      }
    }

    // ---- online softmax: p = exp2((s - m) * cexp) ----
#pragma unroll
    for (int r = 0; r < 8; ++r) {
      float mx = s[0][r];
#pragma unroll
      for (int n = 1; n < 8; ++n) mx = fmaxf(mx, s[n][r]);
#pragma unroll
      for (int msk = 1; msk < 16; msk <<= 1)
        mx = fmaxf(mx, __shfl_xor(mx, msk, 32));
      const float nm    = fmaxf(mrow[r], mx);
      const float alpha = fast_exp2((mrow[r] - nm) * cexp);
      mrow[r] = nm;
      const float nmc = nm * cexp;      // fma-able: s*cexp - nmc
      float sum = 0.0f;
#pragma unroll
      for (int n = 0; n < 8; ++n) {
        const float p = fast_exp2(__builtin_fmaf(s[n][r], cexp, -nmc));
        s[n][r] = p;
        sum += p;
      }
#pragma unroll
      for (int msk = 1; msk < 16; msk <<= 1)
        sum += __shfl_xor(sum, msk, 32);
      lrow[r] = lrow[r] * alpha + sum;
#pragma unroll
      for (int n = 0; n < 8; ++n) o[n][r] *= alpha;
    }

    // ---- P: C-layout -> LDS (row-major bf16) -> A fragments ----
#pragma unroll
    for (int n = 0; n < 8; ++n) {
      const v8bf pb = __builtin_convertvector(s[n], v8bf);
#pragma unroll
      for (int r = 0; r < 8; ++r)
        plds[wv][r + lh * 8][n * 16 + l15] = pb[r];
    }
    // same-wave LDS ops are in-order; no cross-wave sharing here
    v16bf pa[4];
#pragma unroll
    for (int c = 0; c < 4; ++c) {
      const __bf16* pp = &plds[wv][l15][c * 32 + lh * 16];
      Frag16 u;
      u.h[0] = *(const v8bf*)pp;
      u.h[1] = *(const v8bf*)(pp + 8);
      pa[c] = u.v;
    }

    // ---- O += P @ V  (V pre-transposed: contiguous over keys) ----
#pragma unroll
    for (int n = 0; n < 8; ++n) {
      const __bf16* vp = vbat + (size_t)(n * 16 + l15) * TSEQ + jb * 128;
#pragma unroll
      for (int c = 0; c < 4; ++c) {
        // B-fragment = 32 contiguous bytes along the key dimension of vT
        const v16bf bf = *(const v16bf*)(vp + c * 32 + lh * 16);
        o[n] = __builtin_amdgcn_wmma_f32_16x16x32_bf16(
            false, pa[c], false, bf, (short)0, o[n], false, false);
      }
    }
  }

  // ---- normalize and store f32 output ----
#pragma unroll
  for (int n = 0; n < 8; ++n) {
#pragma unroll
    for (int r = 0; r < 8; ++r) {
      const int row = t0 + wv * 16 + r + lh * 8;
      out[((size_t)b * TSEQ + row) * HEAD + n * 16 + l15] = o[n][r] / lrow[r];
    }
  }
}

// ---------------------------------------------------------------------------
extern "C" void kernel_launch(void* const* d_in, const int* in_sizes, int n_in,
                              void* d_out, int out_size, void* d_ws, size_t ws_size,
                              hipStream_t stream) {
  (void)in_sizes; (void)n_in; (void)out_size; (void)ws_size;
  const float* x  = (const float*)d_in[0];
  const float* Wk = (const float*)d_in[1];
  const float* Wq = (const float*)d_in[2];
  const float* Wv = (const float*)d_in[3];
  float* out = (float*)d_out;

  __bf16* qws  = (__bf16*)d_ws;                       // [16384][128] bf16
  __bf16* kws  = qws + (size_t)NROWS * HEAD;          // [16384][128] bf16
  __bf16* vTws = kws + (size_t)NROWS * HEAD;          // [16][128][1024] bf16

  dim3 g1(NROWS / 128, 3), b1(256);
  qkv_proj_kernel<<<g1, b1, 0, stream>>>(x, Wq, Wk, Wv, qws, kws, vTws);

  dim3 g2(TSEQ / 128, BATCH), b2(256);
  attn_kernel<<<g2, b2, 0, stream>>>(qws, kws, vTws, out);
}
